// _DeformConv2d_83734682403409
// MI455X (gfx1250) — compile-verified, matrix operand order checked by reading, not run on hardware
//
#include <hip/hip_runtime.h>
#include <hip/hip_bf16.h>

typedef __attribute__((ext_vector_type(16))) _Float16 v16h;
typedef __attribute__((ext_vector_type(8)))  float    v8f;

#define BATCH   4
#define C_IN    64
#define HH      128
#define WW      128
#define HW      (HH * WW)          // 16384 = 1<<14
#define C_OFF   18                 // 2*K offset channels
#define C_OUT   64
#define CK      576                // C_IN * 9  (GEMM K dim)
#define KCHUNKS 18                 // CK / 32
#define PIX_PER_BLK 32

// ---------------- Kernel 1: offset conv (18ch, 3x3, pad 1) + bias ----------------
__global__ __launch_bounds__(256)
void offset_conv_kernel(const float* __restrict__ x,
                        const float* __restrict__ w_p,
                        const float* __restrict__ b_p,
                        float* __restrict__ offs) {
  int idx = blockIdx.x * blockDim.x + threadIdx.x;   // over B*18*H*W, exact
  int w  = idx & (WW - 1);
  int h  = (idx >> 7) & (HH - 1);
  int oc = (idx >> 14) % C_OFF;
  int b  = idx / (C_OFF << 14);
  float acc = b_p[oc];
  const float* xb = x   + b  * (C_IN * HW);
  const float* wb = w_p + oc * CK;
  for (int ky = 0; ky < 3; ++ky) {
    int yy = h + ky - 1;
    if (yy < 0 || yy >= HH) continue;
    for (int kx = 0; kx < 3; ++kx) {
      int xx = w + kx - 1;
      if (xx < 0 || xx >= WW) continue;
      const float* xp = xb + yy * WW + xx;
      const float* wp = wb + ky * 3 + kx;
      #pragma unroll 8
      for (int c = 0; c < C_IN; ++c)
        acc = fmaf(xp[c * HW], wp[c * 9], acc);
    }
  }
  offs[idx] = acc;   // layout ((b*18 + ch)*H + h)*W + w, ch = k*2 + {y,x}
}

// ---------- Kernel 2: pack w_d -> f16 in WMMA 16-bit A (16x32) fragment layout ----------
// Fragment element e (0..15) of lane l: r=e>>1, q=e&1,
//   k_local = (r<4 ? 0 : 16) + 8*(l>>4) + (r&3)*2 + q ;  row M = (l&15)
__global__ void pack_weights_kernel(const float* __restrict__ w_d,
                                    _Float16* __restrict__ wf16) {
  int idx = blockIdx.x * blockDim.x + threadIdx.x;   // over 4*18*32*16 = 36864, exact
  int e    = idx & 15;
  int lane = (idx >> 4) & 31;
  int kc   = (idx >> 9) % KCHUNKS;
  int mt   = idx / (KCHUNKS * 512);
  int m    = mt * 16 + (lane & 15);
  int r = e >> 1, q = e & 1;
  int kl = ((r < 4) ? 0 : 16) + ((lane >> 4) << 3) + (r & 3) * 2 + q;
  int kk = kc * 32 + kl;                              // kk = c*9 + tap
  wf16[idx] = (_Float16)w_d[m * CK + kk];
}

// ---------------- Kernel 3: fused bilinear sampling + WMMA GEMM ----------------
// LDS holds the sampled tile ALREADY in B-fragment order so each wave reads its
// whole 32-byte fragment with one aligned v16h load (2x ds_load_b128) per WMMA.
// Inverse fragment map for k_local kl in [0,32):
//   lane half h = (kl>>3)&1 ;  element e = (kl&7) + (kl>=16 ? 8 : 0)
// Fragment address: ldsB[((nt*KCHUNKS + kc)*32 + h*16 + (pix&15))*16 + e]
__global__ __launch_bounds__(256)
void deform_wmma_kernel(const float* __restrict__ x,
                        const float* __restrict__ offs,
                        const _Float16* __restrict__ wf16,
                        float* __restrict__ out) {
  __shared__ __align__(32) _Float16 ldsB[2 * KCHUNKS * 32 * 16];  // 36 KB
  const int tid = threadIdx.x;
  const int pixBase = blockIdx.x * PIX_PER_BLK;

  // ---- Stage bilinear samples for 32 pixels x 9 taps x 64 channels into LDS ----
  for (int pt = tid; pt < PIX_PER_BLK * 9; pt += 256) {
    int pix = pt / 9;
    int kr  = pt - pix * 9;
    int gp  = pixBase + pix;
    int b = gp >> 14, h = (gp >> 7) & 127, w = gp & 127;
    const float* ob = offs + ((b * C_OFF) << 14) + (h << 7) + w;
    float py = ob[(size_t)(kr * 2 + 0) << 14] + (float)(h - 1 + kr / 3);
    float px = ob[(size_t)(kr * 2 + 1) << 14] + (float)(w - 1 + kr % 3);
    float y0f = floorf(py), x0f = floorf(px);
    int y0 = (int)y0f, x0 = (int)x0f;
    int y1 = y0 + 1,   x1 = x0 + 1;
    float wy = py - y0f, wx = px - x0f;
    float m00 = ((y0 >= 0) & (y0 < HH) & (x0 >= 0) & (x0 < WW)) ? (1.f - wy) * (1.f - wx) : 0.f;
    float m01 = ((y0 >= 0) & (y0 < HH) & (x1 >= 0) & (x1 < WW)) ? (1.f - wy) * wx         : 0.f;
    float m10 = ((y1 >= 0) & (y1 < HH) & (x0 >= 0) & (x0 < WW)) ? wy * (1.f - wx)         : 0.f;
    float m11 = ((y1 >= 0) & (y1 < HH) & (x1 >= 0) & (x1 < WW)) ? wy * wx                 : 0.f;
    int y0c = min(max(y0, 0), HH - 1), x0c = min(max(x0, 0), WW - 1);
    int y1c = min(max(y1, 0), HH - 1), x1c = min(max(x1, 0), WW - 1);
    int i00 = y0c * WW + x0c, i01 = y0c * WW + x1c;
    int i10 = y1c * WW + x0c, i11 = y1c * WW + x1c;
    const float* xb = x + b * (C_IN * HW);
    const int nt = pix >> 4;
    const int lnLo = pix & 15;
    for (int c = 0; c < C_IN; ++c) {
      const float* xc = xb + c * HW;
      float s = m00 * xc[i00] + m01 * xc[i01] + m10 * xc[i10] + m11 * xc[i11];
      int kk = c * 9 + kr;
      int kc = kk >> 5, kl = kk & 31;
      int hhalf = (kl >> 3) & 1;
      int e = (kl & 7) + ((kl & 16) >> 1);           // +8 when kl>=16
      ldsB[(((nt * KCHUNKS + kc) * 32) + (hhalf << 4) + lnLo) * 16 + e] = (_Float16)s;
    }
  }
  __syncthreads();

  // ---- WMMA: 8 waves = 4 M-tiles x 2 N-tiles; K-loop 18 chunks of 32 ----
  const int wave = tid >> 5, lane = tid & 31;
  const int mt = wave >> 1, nt = wave & 1;
  const _Float16* aBase = wf16 + (size_t)(mt * KCHUNKS) * 32 * 16 + lane * 16;
  const _Float16* bBase = ldsB + (size_t)(nt * KCHUNKS) * 32 * 16 + lane * 16;
  v8f acc = {};
  for (int kc = 0; kc < KCHUNKS; ++kc) {
    v16h a  = *(const v16h*)(aBase + (size_t)kc * 32 * 16);
    v16h bf = *(const v16h*)(bBase + (size_t)kc * 32 * 16);
    acc = __builtin_amdgcn_wmma_f32_16x16x32_f16(
        /*neg_a=*/false, a, /*neg_b=*/false, bf,
        /*c_mod=*/(short)0, acc, /*reuse_a=*/false, /*reuse_b=*/false);
  }

  // ---- Store D: VGPR i -> M = mt*16 + i + 8*(lane>=16), N = lane&15 ----
  const int hh8 = (lane >> 4) << 3;
  int n = (nt << 4) + (lane & 15);
  int gp = pixBase + n;
  int b = gp >> 14, h = (gp >> 7) & 127, w = gp & 127;
  int mbase = (mt << 4) + hh8;
  float* op = out + (size_t)((b * C_OUT + mbase) << 14) + (h << 7) + w;
  #pragma unroll
  for (int i = 0; i < 8; ++i)
    op[(size_t)i << 14] = acc[i];
}

extern "C" void kernel_launch(void* const* d_in, const int* in_sizes, int n_in,
                              void* d_out, int out_size, void* d_ws, size_t ws_size,
                              hipStream_t stream) {
  (void)in_sizes; (void)n_in; (void)out_size; (void)ws_size;
  const float* x   = (const float*)d_in[0];
  const float* w_p = (const float*)d_in[1];
  const float* b_p = (const float*)d_in[2];
  const float* w_d = (const float*)d_in[3];
  float* out = (float*)d_out;

  // workspace partition: [offsets: B*18*H*W f32][packed f16 weights: 4*18*32*16]
  float*     offs = (float*)d_ws;
  _Float16*  wf16 = (_Float16*)((char*)d_ws + (size_t)BATCH * C_OFF * HW * sizeof(float));

  offset_conv_kernel<<<(BATCH * C_OFF * HW) / 256, 256, 0, stream>>>(x, w_p, b_p, offs);
  pack_weights_kernel<<<(4 * KCHUNKS * 32 * 16) / 256, 256, 0, stream>>>(w_d, wf16);
  deform_wmma_kernel<<<(BATCH * HW) / PIX_PER_BLK, 256, 0, stream>>>(x, offs, wf16, out);
}
